// KVAE_34445637714558
// MI455X (gfx1250) — compile-verified, matrix operand order checked by reading, not run on hardware
//
#include <hip/hip_runtime.h>
#include <hip/hip_bf16.h>
#include <math.h>

typedef __attribute__((ext_vector_type(16))) _Float16 v16h;
typedef __attribute__((ext_vector_type(8)))  _Float16 v8h;
typedef __attribute__((ext_vector_type(8)))  float    v8f;

#define DEVINL __device__ __forceinline__

union V16H { v16h v; v8h h[2]; };

// ---- WMMA helpers (gfx1250: D = A(16x32 f16) x B(32x16 f16) + C(16x16 f32)) ----
DEVINL v8f wmma_f16(v16h a, v16h b, v8f c) {
  return __builtin_amdgcn_wmma_f32_16x16x32_f16(false, a, false, b, (short)0, c,
                                                false, false);
}

// A operand: row-major 16x128 f16 tile (LDS). ISA layout: lane (m=l&15, hi=l>>4);
// halves 0..7  = k in [kt*32 + hi*8,      +8)
// halves 8..15 = k in [kt*32 + 16 + hi*8, +8)
DEVINL v16h ld_a16(const _Float16* base, int lane, int kt) {
  int m = lane & 15, hi = lane >> 4;
  const _Float16* p = base + m * 128 + kt * 32 + hi * 8;
  V16H u;
  u.h[0] = *(const v8h*)p;
  u.h[1] = *(const v8h*)(p + 16);
  return u.v;
}

// B operand: W stored row-major [N][K=128] (i.e. we compute h @ W^T).
// Mirrored pattern: lane n = l&15 selects row, same k-chunking as A.
DEVINL v16h ld_b16(const _Float16* W, int row_base, int lane, int kt) {
  int n = lane & 15, hi = lane >> 4;
  const _Float16* p = W + (row_base + n) * 128 + kt * 32 + hi * 8;
  V16H u;
  u.h[0] = *(const v8h*)p;
  u.h[1] = *(const v8h*)(p + 16);
  return u.v;
}

// ------------------------------------------------------------------
// Prep: fp32 -> fp16 weight conversion + accumulator zeroing
// ------------------------------------------------------------------
__global__ __launch_bounds__(256) void prep_kernel(
    const float* __restrict__ whh, const float* __restrict__ ew2,
    const float* __restrict__ dw2,
    _Float16* whh_h, _Float16* ew2h, _Float16* dw2h,
    float* recon_acc, float* ent_acc) {
  int i = blockIdx.x * 256 + threadIdx.x;
  if (i < 65536) whh_h[i] = (_Float16)whh[i];
  if (i < 16384) { ew2h[i] = (_Float16)ew2[i]; dw2h[i] = (_Float16)dw2[i]; }
  if (i < 512)   { recon_acc[i] = 0.f; ent_acc[i] = 0.f; }
}

// ------------------------------------------------------------------
// Fused encoder + reparam + decoder + loss partials.
// 128 threads = 4 waves; each wave owns a 16-token tile (64 tokens/block).
// Layer2 (128x128) runs on WMMA; layers 1/3 (K=2 / N<=4) on VALU.
// ------------------------------------------------------------------
__global__ __launch_bounds__(128) void vae_kernel(
    const float* __restrict__ x, const float* __restrict__ eps,
    const float* __restrict__ ew1, const float* __restrict__ eb1,
    const _Float16* __restrict__ ew2h, const float* __restrict__ eb2,
    const float* __restrict__ ew3, const float* __restrict__ eb3,
    const float* __restrict__ dw1, const float* __restrict__ db1,
    const _Float16* __restrict__ dw2h, const float* __restrict__ db2,
    const float* __restrict__ dw3, const float* __restrict__ db3,
    float* __restrict__ a_out, float* ent_acc, float* recon_acc) {
  __shared__ __align__(16) _Float16 slab1[4][2048];  // per-wave 16x128
  __shared__ __align__(16) _Float16 slab2[4][2048];
  __shared__ float xs[4][32];
  __shared__ float outb[4][16][4];
  __shared__ float abuf[4][32];
  __shared__ float wsum[4];

  const int tid = threadIdx.x, w = tid >> 5, lane = tid & 31;
  const int tokbase = blockIdx.x * 64 + w * 16;  // 16 tokens per wave
  const int b = tokbase >> 10;                   // T = 1024, tile stays in one b
  const int hi = lane >> 4;

  // stage scaled x, zero wave accumulator
  xs[w][lane] = x[tokbase * 2 + lane] * (1.0f / 32.0f);
  if (lane == 0) wsum[w] = 0.f;
  __syncthreads();

  // encoder layer1: h1 = relu(x @ W1^T + b1), K=2
  for (int i = lane; i < 2048; i += 32) {
    int m = i >> 7, j = i & 127;
    float v = xs[w][m * 2] * ew1[j * 2] + xs[w][m * 2 + 1] * ew1[j * 2 + 1] + eb1[j];
    slab1[w][m * 128 + j] = (_Float16)(v > 0.f ? v : 0.f);
  }
  __syncthreads();

  // encoder layer2: h2 = relu(h1 @ W2^T + b2)  [WMMA]
#pragma unroll
  for (int nt = 0; nt < 8; ++nt) {
    v8f acc = {};
#pragma unroll
    for (int kt = 0; kt < 4; ++kt)
      acc = wmma_f16(ld_a16(slab1[w], lane, kt), ld_b16(ew2h, nt * 16, lane, kt), acc);
    int n = nt * 16 + (lane & 15);
    float bias = eb2[n];
#pragma unroll
    for (int r = 0; r < 8; ++r) {
      float v = acc[r] + bias;
      slab2[w][(r + 8 * hi) * 128 + n] = (_Float16)(v > 0.f ? v : 0.f);
    }
  }
  __syncthreads();

  // encoder layer3: 4 outputs per token
#pragma unroll
  for (int pp = 0; pp < 2; ++pp) {
    int p = lane + pp * 32, m = p >> 2, o = p & 3;
    float s = eb3[o];
    for (int j = 0; j < 128; ++j) s += (float)slab2[w][m * 128 + j] * ew3[o * 128 + j];
    outb[w][m][o] = s;
  }
  __syncthreads();

  // reparameterize: a = mu + eps*exp(0.5*logvar); entropy partial
  {
    int m = lane >> 1, ad = lane & 1;
    float mu = outb[w][m][ad];
    float lv = outb[w][m][2 + ad];
    float a = mu + eps[tokbase * 2 + lane] * __expf(0.5f * lv);
    abuf[w][lane] = a;
    a_out[tokbase * 2 + lane] = a;
    atomicAdd(&wsum[w], 0.5f * (1.f + lv));
  }
  __syncthreads();

  if (lane == 0) { atomicAdd(&ent_acc[b], wsum[w]); wsum[w] = 0.f; }
  // decoder layer1 from abuf
  for (int i = lane; i < 2048; i += 32) {
    int m = i >> 7, j = i & 127;
    float v = abuf[w][m * 2] * dw1[j * 2] + abuf[w][m * 2 + 1] * dw1[j * 2 + 1] + db1[j];
    slab1[w][m * 128 + j] = (_Float16)(v > 0.f ? v : 0.f);
  }
  __syncthreads();

  // decoder layer2 [WMMA]
#pragma unroll
  for (int nt = 0; nt < 8; ++nt) {
    v8f acc = {};
#pragma unroll
    for (int kt = 0; kt < 4; ++kt)
      acc = wmma_f16(ld_a16(slab1[w], lane, kt), ld_b16(dw2h, nt * 16, lane, kt), acc);
    int n = nt * 16 + (lane & 15);
    float bias = db2[n];
#pragma unroll
    for (int r = 0; r < 8; ++r) {
      float v = acc[r] + bias;
      slab2[w][(r + 8 * hi) * 128 + n] = (_Float16)(v > 0.f ? v : 0.f);
    }
  }
  __syncthreads();

  // decoder layer3 + squared error vs x (unscaled)
  {
    int m = lane >> 1, o = lane & 1;
    float s = db3[o];
    for (int j = 0; j < 128; ++j) s += (float)slab2[w][m * 128 + j] * dw3[o * 128 + j];
    float xr = s * 32.0f;
    float d = x[tokbase * 2 + lane] - xr;
    atomicAdd(&wsum[w], d * d);
  }
  __syncthreads();
  if (lane == 0) atomicAdd(&recon_acc[b], wsum[w]);
}

// ------------------------------------------------------------------
// LSTM over T=1024. 32 blocks x 256 threads; block = 16 batch rows.
// Each wave holds its W_hh slice (16 v16h) in VGPRs for all steps;
// h lives in a 16x128 f16 LDS tile. Alpha (softmax, K=3) computed in-kernel.
// ------------------------------------------------------------------
__global__ __launch_bounds__(256) void lstm_kernel(
    const float* __restrict__ a_sample, const _Float16* __restrict__ whh_h,
    const float* __restrict__ wih, const float* __restrict__ b_ih,
    const float* __restrict__ b_hh, const float* __restrict__ alpha_w,
    const float* __restrict__ alpha_b, float* __restrict__ alpha_out) {
  __shared__ __align__(16) _Float16 hbuf[16 * 128];
  __shared__ float xbuf[32];
  __shared__ float sc[48];

  const int tid = threadIdx.x, w = tid >> 5, lane = tid & 31;
  const int mb = blockIdx.x * 16;
  const int col = w * 16 + (lane & 15);
  const int hi = lane >> 4;

  // Preload W_hh slice: 4 gates x 4 k-tiles, rows [g*128 + w*16, +16)
  v16h Bw[16];
#pragma unroll
  for (int g = 0; g < 4; ++g)
#pragma unroll
    for (int kt = 0; kt < 4; ++kt)
      Bw[g * 4 + kt] = ld_b16(whh_h, g * 128 + w * 16, lane, kt);

  float wih0[4], wih1[4], bias[4];
#pragma unroll
  for (int g = 0; g < 4; ++g) {
    int row = g * 128 + col;
    wih0[g] = wih[row * 2 + 0];
    wih1[g] = wih[row * 2 + 1];
    bias[g] = b_ih[row] + b_hh[row];
  }

  for (int i = tid; i < 2048; i += 256) hbuf[i] = (_Float16)0.f;
  v8f cst = {};

  for (int t = 0; t < 1024; ++t) {
    if (tid < 32) {
      int m = tid >> 1, cd = tid & 1;
      xbuf[tid] = (t == 0) ? 0.f : a_sample[((mb + m) * 1024 + (t - 1)) * 2 + cd];
    }
    __syncthreads();  // xbuf ready, hbuf(t-1) ready

    v8f a0 = {}, a1 = {}, a2 = {}, a3 = {};
#pragma unroll
    for (int kt = 0; kt < 4; ++kt) {
      v16h a = ld_a16(hbuf, lane, kt);
      a0 = wmma_f16(a, Bw[0 * 4 + kt], a0);
      a1 = wmma_f16(a, Bw[1 * 4 + kt], a1);
      a2 = wmma_f16(a, Bw[2 * 4 + kt], a2);
      a3 = wmma_f16(a, Bw[3 * 4 + kt], a3);
    }

    float hreg[8];
#pragma unroll
    for (int r = 0; r < 8; ++r) {
      int m = r + 8 * hi;
      float x0 = xbuf[m * 2], x1 = xbuf[m * 2 + 1];
      float gi = a0[r] + x0 * wih0[0] + x1 * wih1[0] + bias[0];
      float gf = a1[r] + x0 * wih0[1] + x1 * wih1[1] + bias[1];
      float gg = a2[r] + x0 * wih0[2] + x1 * wih1[2] + bias[2];
      float go = a3[r] + x0 * wih0[3] + x1 * wih1[3] + bias[3];
      float si = 1.f / (1.f + __expf(-gi));
      float sf = 1.f / (1.f + __expf(-gf));
      float so = 1.f / (1.f + __expf(-go));
      float c = sf * cst[r] + si * tanhf(gg);
      cst[r] = c;
      hreg[r] = so * tanhf(c);
    }
    __syncthreads();  // all waves done reading hbuf(t-1)

#pragma unroll
    for (int r = 0; r < 8; ++r)
      hbuf[(r + 8 * hi) * 128 + col] = (_Float16)hreg[r];
    __syncthreads();  // hbuf(t) ready

    // alpha = softmax(h @ alpha_w^T + alpha_b), K=3
    if (tid < 48) {
      int m = tid / 3, k = tid % 3;
      float s = alpha_b[k];
      for (int j = 0; j < 128; ++j) s += (float)hbuf[m * 128 + j] * alpha_w[k * 128 + j];
      sc[tid] = s;
    }
    __syncthreads();
    if (tid < 16) {
      int m = tid;
      float s0 = sc[m * 3], s1 = sc[m * 3 + 1], s2 = sc[m * 3 + 2];
      float mx = fmaxf(s0, fmaxf(s1, s2));
      float e0 = __expf(s0 - mx), e1 = __expf(s1 - mx), e2 = __expf(s2 - mx);
      float inv = 1.f / (e0 + e1 + e2);
      int base = (t * 512 + mb + m) * 3;
      alpha_out[base + 0] = e0 * inv;
      alpha_out[base + 1] = e1 * inv;
      alpha_out[base + 2] = e2 * inv;
    }
  }
}

// ------------------------------------------------------------------
// Kalman filter: one thread per batch chain (Z=4, A=2), analytic 2x2 inverse.
// ------------------------------------------------------------------
__global__ __launch_bounds__(256) void kf_kernel(
    const float* __restrict__ a_sample, const float* __restrict__ alpha,
    const float* __restrict__ Ak, const float* __restrict__ Bk,
    const float* __restrict__ Ck, const float* __restrict__ Dk,
    const float* __restrict__ Q_lv, const float* __restrict__ R_lv,
    const float* __restrict__ z0mu, const float* __restrict__ z0lv,
    float* __restrict__ ll_out) {
  int b = blockIdx.x * 256 + threadIdx.x;
  if (b >= 512) return;

  float Akr[3][16], Bkr[3][4], Ckr[3][8], Dkr[3][2], Qd[4], Rd[2];
#pragma unroll
  for (int k = 0; k < 3; ++k) {
#pragma unroll
    for (int i = 0; i < 16; ++i) Akr[k][i] = Ak[k * 16 + i];
#pragma unroll
    for (int i = 0; i < 4; ++i) Bkr[k][i] = Bk[k * 4 + i];
#pragma unroll
    for (int i = 0; i < 8; ++i) Ckr[k][i] = Ck[k * 8 + i];
#pragma unroll
    for (int i = 0; i < 2; ++i) Dkr[k][i] = Dk[k * 2 + i];
  }
#pragma unroll
  for (int i = 0; i < 4; ++i) Qd[i] = __expf(Q_lv[i]);
#pragma unroll
  for (int i = 0; i < 2; ++i) Rd[i] = __expf(R_lv[i]);

  float mu[4], Sg[16];
#pragma unroll
  for (int i = 0; i < 4; ++i) mu[i] = z0mu[i];
#pragma unroll
  for (int i = 0; i < 16; ++i) Sg[i] = 0.f;
#pragma unroll
  for (int i = 0; i < 4; ++i) Sg[i * 4 + i] = __expf(z0lv[i]);

  float ll = 0.f;
  const float LOG2PI = 1.8378770664093453f;

  for (int t = 0; t < 1024; ++t) {
    int ab = (t * 512 + b) * 3;
    float al0 = alpha[ab], al1 = alpha[ab + 1], al2 = alpha[ab + 2];
    float At[16], Bt[4], Ct[8], Dt[2];
#pragma unroll
    for (int i = 0; i < 16; ++i) At[i] = al0 * Akr[0][i] + al1 * Akr[1][i] + al2 * Akr[2][i];
#pragma unroll
    for (int i = 0; i < 4; ++i) Bt[i] = al0 * Bkr[0][i] + al1 * Bkr[1][i] + al2 * Bkr[2][i];
#pragma unroll
    for (int i = 0; i < 8; ++i) Ct[i] = al0 * Ckr[0][i] + al1 * Ckr[1][i] + al2 * Ckr[2][i];
#pragma unroll
    for (int i = 0; i < 2; ++i) Dt[i] = al0 * Dkr[0][i] + al1 * Dkr[1][i] + al2 * Dkr[2][i];

    float mp[4];
#pragma unroll
    for (int i = 0; i < 4; ++i) {
      float s = Bt[i];
#pragma unroll
      for (int j = 0; j < 4; ++j) s += At[i * 4 + j] * mu[j];
      mp[i] = s;
    }
    float AS[16];
#pragma unroll
    for (int i = 0; i < 4; ++i)
#pragma unroll
      for (int j = 0; j < 4; ++j) {
        float s = 0.f;
#pragma unroll
        for (int k = 0; k < 4; ++k) s += At[i * 4 + k] * Sg[k * 4 + j];
        AS[i * 4 + j] = s;
      }
    float Sp[16];
#pragma unroll
    for (int i = 0; i < 4; ++i)
#pragma unroll
      for (int l = 0; l < 4; ++l) {
        float s = (i == l) ? Qd[i] : 0.f;
#pragma unroll
        for (int k = 0; k < 4; ++k) s += AS[i * 4 + k] * At[l * 4 + k];
        Sp[i * 4 + l] = s;
      }

    float ao0 = a_sample[(b * 1024 + t) * 2 + 0];
    float ao1 = a_sample[(b * 1024 + t) * 2 + 1];
    float y0 = Dt[0], y1 = Dt[1];
#pragma unroll
    for (int j = 0; j < 4; ++j) { y0 += Ct[j] * mp[j]; y1 += Ct[4 + j] * mp[j]; }
    float r0 = ao0 - y0, r1 = ao1 - y1;

    float CS[8];
#pragma unroll
    for (int a = 0; a < 2; ++a)
#pragma unroll
      for (int j = 0; j < 4; ++j) {
        float s = 0.f;
#pragma unroll
        for (int k = 0; k < 4; ++k) s += Ct[a * 4 + k] * Sp[k * 4 + j];
        CS[a * 4 + j] = s;
      }
    float S00 = Rd[0], S01 = 0.f, S10 = 0.f, S11 = Rd[1];
#pragma unroll
    for (int k = 0; k < 4; ++k) {
      S00 += CS[k] * Ct[k];
      S01 += CS[k] * Ct[4 + k];
      S10 += CS[4 + k] * Ct[k];
      S11 += CS[4 + k] * Ct[4 + k];
    }
    float det = S00 * S11 - S01 * S10;
    float idet = 1.f / det;
    float i00 = S11 * idet, i01 = -S01 * idet, i10 = -S10 * idet, i11 = S00 * idet;

    float PC[8];
#pragma unroll
    for (int i = 0; i < 4; ++i) {
      float p0 = 0.f, p1 = 0.f;
#pragma unroll
      for (int k = 0; k < 4; ++k) { p0 += Sp[i * 4 + k] * Ct[k]; p1 += Sp[i * 4 + k] * Ct[4 + k]; }
      PC[i * 2 + 0] = p0; PC[i * 2 + 1] = p1;
    }
    float Kg[8];
#pragma unroll
    for (int i = 0; i < 4; ++i) {
      Kg[i * 2 + 0] = PC[i * 2] * i00 + PC[i * 2 + 1] * i10;
      Kg[i * 2 + 1] = PC[i * 2] * i01 + PC[i * 2 + 1] * i11;
    }
#pragma unroll
    for (int i = 0; i < 4; ++i) mu[i] = mp[i] + Kg[i * 2] * r0 + Kg[i * 2 + 1] * r1;

    float KC[16];
#pragma unroll
    for (int i = 0; i < 4; ++i)
#pragma unroll
      for (int j = 0; j < 4; ++j)
        KC[i * 4 + j] = Kg[i * 2] * Ct[j] + Kg[i * 2 + 1] * Ct[4 + j];
#pragma unroll
    for (int i = 0; i < 4; ++i)
#pragma unroll
      for (int j = 0; j < 4; ++j) {
        float s = Sp[i * 4 + j];
#pragma unroll
        for (int k = 0; k < 4; ++k) s -= KC[i * 4 + k] * Sp[k * 4 + j];
        Sg[i * 4 + j] = s;  // safe: Sg fully overwritten, read only via Sp
      }

    float quad = r0 * (i00 * r0 + i01 * r1) + r1 * (i10 * r0 + i11 * r1);
    ll -= 0.5f * (__logf(det) + quad + 2.f * LOG2PI);
  }
  ll_out[b] = ll;
}

// ------------------------------------------------------------------
// Final reduction over B to 3 scalars
// ------------------------------------------------------------------
__global__ __launch_bounds__(256) void final_kernel(
    const float* __restrict__ recon_acc, const float* __restrict__ ent_acc,
    const float* __restrict__ ll, const float* __restrict__ log_scale_x,
    float* __restrict__ out) {
  __shared__ float s0[256], s1[256], s2[256];
  int tid = threadIdx.x;
  float var_x = __expf(2.f * log_scale_x[0]);
  float sa = 0.f, sb = 0.f, sc2 = 0.f;
  for (int b = tid; b < 512; b += 256) {
    float rl = 0.5f * recon_acc[b] / var_x;
    float elbo = -rl + ll[b] + ent_acc[b];
    sa += -elbo;
    sb += rl;
    sc2 += -ll[b];
  }
  s0[tid] = sa; s1[tid] = sb; s2[tid] = sc2;
  __syncthreads();
  for (int st = 128; st > 0; st >>= 1) {
    if (tid < st) { s0[tid] += s0[tid + st]; s1[tid] += s1[tid + st]; s2[tid] += s2[tid + st]; }
    __syncthreads();
  }
  if (tid == 0) {
    out[0] = s0[0] * (1.f / 512.f);
    out[1] = s1[0] * (1.f / 512.f);
    out[2] = s2[0] * (1.f / 512.f);
  }
}

extern "C" void kernel_launch(void* const* d_in, const int* in_sizes, int n_in,
                              void* d_out, int out_size, void* d_ws, size_t ws_size,
                              hipStream_t stream) {
  const float* x        = (const float*)d_in[0];
  const float* eps_a    = (const float*)d_in[1];
  const float* enc_w1   = (const float*)d_in[2];
  const float* enc_b1   = (const float*)d_in[3];
  const float* enc_w2   = (const float*)d_in[4];
  const float* enc_b2   = (const float*)d_in[5];
  const float* enc_w3   = (const float*)d_in[6];
  const float* enc_b3   = (const float*)d_in[7];
  const float* dec_w1   = (const float*)d_in[8];
  const float* dec_b1   = (const float*)d_in[9];
  const float* dec_w2   = (const float*)d_in[10];
  const float* dec_b2   = (const float*)d_in[11];
  const float* dec_w3   = (const float*)d_in[12];
  const float* dec_b3   = (const float*)d_in[13];
  const float* W_ih     = (const float*)d_in[14];
  const float* W_hh     = (const float*)d_in[15];
  const float* b_ih     = (const float*)d_in[16];
  const float* b_hh     = (const float*)d_in[17];
  const float* alpha_w  = (const float*)d_in[18];
  const float* alpha_b  = (const float*)d_in[19];
  const float* Ak       = (const float*)d_in[20];
  const float* Bk       = (const float*)d_in[21];
  const float* Ck       = (const float*)d_in[22];
  const float* Dk       = (const float*)d_in[23];
  const float* Q_lv     = (const float*)d_in[24];
  const float* R_lv     = (const float*)d_in[25];
  const float* z0mu     = (const float*)d_in[26];
  const float* z0lv     = (const float*)d_in[27];
  const float* log_sx   = (const float*)d_in[28];

  char* w = (char*)d_ws;
  float*    a_sample  = (float*)(w);                  // 512*1024*2 f32 = 4 MiB
  float*    alpha     = (float*)(w + 4194304);        // 1024*512*3 f32 = 6 MiB
  _Float16* whh_h     = (_Float16*)(w + 10485760);    // 512*128 f16
  _Float16* ew2h      = (_Float16*)(w + 10616832);    // 128*128 f16
  _Float16* dw2h      = (_Float16*)(w + 10649600);    // 128*128 f16
  float*    recon_acc = (float*)(w + 10682368);       // 512 f32
  float*    ent_acc   = (float*)(w + 10684416);       // 512 f32
  float*    ll        = (float*)(w + 10686464);       // 512 f32
  float*    out       = (float*)d_out;                // 3 f32

  prep_kernel<<<256, 256, 0, stream>>>(W_hh, enc_w2, dec_w2, whh_h, ew2h, dw2h,
                                       recon_acc, ent_acc);

  // 524288 tokens / 64 per block
  vae_kernel<<<8192, 128, 0, stream>>>(x, eps_a, enc_w1, enc_b1, ew2h, enc_b2,
                                       enc_w3, enc_b3, dec_w1, dec_b1, dw2h,
                                       dec_b2, dec_w3, dec_b3, a_sample,
                                       ent_acc, recon_acc);

  lstm_kernel<<<32, 256, 0, stream>>>(a_sample, whh_h, W_ih, b_ih, b_hh,
                                      alpha_w, alpha_b, alpha);

  kf_kernel<<<2, 256, 0, stream>>>(a_sample, alpha, Ak, Bk, Ck, Dk, Q_lv, R_lv,
                                   z0mu, z0lv, ll);

  final_kernel<<<1, 256, 0, stream>>>(recon_acc, ent_acc, ll, log_sx, out);
}
